// ATGRUEncoder_79783312490662
// MI455X (gfx1250) — matrix operand order repaired and checked by reading, not compiled, not run on hardware
//
#include <hip/hip_runtime.h>
#include <hip/hip_bf16.h>

// ---------------------------------------------------------------------------
// AGCRN 2-layer graph GRU for MI455X (gfx1250, wave32, WMMA).
// Heavy math (A@inp diffusion, per-node adaptive weight apply) runs on
// v_wmma_f32_16x16x32_f16.  A (2048x2048 f16 = 8.4MB) is L2-resident and
// reused 16*B*2*2 times -> compute bound.  Per-node weights materialized
// once in f16 workspace, pre-swizzled into WMMA B-fragment layout so each
// lane loads its fragment slice as one contiguous 32B vector.
// ---------------------------------------------------------------------------

typedef __attribute__((ext_vector_type(16))) _Float16 v16h;
typedef __attribute__((ext_vector_type(8)))  _Float16 v8h;
typedef __attribute__((ext_vector_type(8)))  float    v8f;

#define NN   2048
#define DD   10
#define HH   64
#define BB   8
#define TT   16

// ---- WMMA fragment helpers (wave32; this code's k = 16*(lane/16) + s order, used by A and B alike) ----

// A-matrix 16x32 f16 from row-major tile (ld halves): lane<16 -> row M=lane,
// halves[0..15]=K0..15; lane>=16 -> K16..31.
__device__ __forceinline__ v16h frag_a(const _Float16* p, int ld) {
  const int lane = threadIdx.x & 31;
  const int m  = lane & 15;
  const int kb = (lane >> 4) * 16;
  const _Float16* q = p + (long)m * ld;
  v8h lo = *(const v8h*)(q + kb);
  v8h hi = *(const v8h*)(q + 8 + kb);
  v16h a;
#pragma unroll
  for (int i = 0; i < 8; ++i) { a[i] = lo[i]; a[8 + i] = hi[i]; }
  return a;
}

// Same as frag_a but rows 8..15 are zero (batch M=8 padded to 16).
__device__ __forceinline__ v16h frag_a_zpad8(const _Float16* p, int ld) {
  const int lane = threadIdx.x & 31;
  const int m  = lane & 15;
  const int kb = (lane >> 4) * 16;
  v16h a = {};
  if (m < 8) {
    const _Float16* q = p + (long)m * ld;
    v8h lo = *(const v8h*)(q + kb);
    v8h hi = *(const v8h*)(q + 8 + kb);
#pragma unroll
    for (int i = 0; i < 8; ++i) { a[i] = lo[i]; a[8 + i] = hi[i]; }
  }
  return a;
}

__device__ __forceinline__ v8f wmma_f16(v16h a, v16h b, v8f c) {
  return __builtin_amdgcn_wmma_f32_16x16x32_f16(false, a, false, b, (short)0, c,
                                                false, false);
}

// ---------------------------------------------------------------------------
// Setup kernels
// ---------------------------------------------------------------------------

// A = softmax(relu(E E^T), axis=1), stored f16 row-major.
__global__ void compute_A_kernel(const float* __restrict__ emb,
                                 _Float16* __restrict__ A16) {
  __shared__ float erow[DD];
  __shared__ float red[256];
  const int n = blockIdx.x, tid = threadIdx.x;
  if (tid < DD) erow[tid] = emb[n * DD + tid];
  __syncthreads();
  float vals[NN / 256];
  float mx = -1e30f;
#pragma unroll
  for (int j8 = 0; j8 < NN / 256; ++j8) {
    const int j = tid + j8 * 256;
    float s = 0.f;
#pragma unroll
    for (int d = 0; d < DD; ++d) s += erow[d] * emb[j * DD + d];
    s = fmaxf(s, 0.f);
    vals[j8] = s;
    mx = fmaxf(mx, s);
  }
  red[tid] = mx; __syncthreads();
  for (int s = 128; s > 0; s >>= 1) {
    if (tid < s) red[tid] = fmaxf(red[tid], red[tid + s]);
    __syncthreads();
  }
  mx = red[0]; __syncthreads();
  float sum = 0.f;
#pragma unroll
  for (int j8 = 0; j8 < NN / 256; ++j8) { vals[j8] = expf(vals[j8] - mx); sum += vals[j8]; }
  red[tid] = sum; __syncthreads();
  for (int s = 128; s > 0; s >>= 1) {
    if (tid < s) red[tid] += red[tid + s];
    __syncthreads();
  }
  const float inv = 1.0f / red[0];
#pragma unroll
  for (int j8 = 0; j8 < NN / 256; ++j8)
    A16[(long)n * NN + tid + j8 * 256] = (_Float16)(vals[j8] * inv);
}

// Effective per-node weights, stored in WMMA B-fragment-packed layout:
//   W[(((n*nkc + kc)*nct + ct)*32 + lane)*16 + h]
// where fragment (kc,ct) covers K rows [kc*32,kc*32+32), cols [ct*16,ct*16+16),
// lane = (kb<<4)|c holds B[kc*32 + kb*16 + h][ct*16 + c].
__global__ void weff_kernel(const float* __restrict__ emb,
                            const float* __restrict__ wpool,
                            _Float16* __restrict__ W,
                            int Ireal, int Ipad, int O, int KI) {
  const long idx = (long)blockIdx.x * 256 + threadIdx.x;
  const long total = (long)NN * KI * O;
  if (idx >= total) return;
  const int h = (int)(idx & 15);
  long t = idx >> 4;
  const int l = (int)(t & 31);
  t >>= 5;
  const int nct = O >> 4;
  const int nkc = KI >> 5;
  const int ct = (int)(t % nct); t /= nct;
  const int kc = (int)(t % nkc);
  const int n  = (int)(t / nkc);
  const int ki = kc * 32 + (l >> 4) * 16 + h;   // flattened (k,i)
  const int o  = ct * 16 + (l & 15);
  const int k = ki / Ipad, i = ki % Ipad;
  float acc = 0.f;
  if (i < Ireal) {
#pragma unroll
    for (int d = 0; d < DD; ++d)
      acc += emb[n * DD + d] * wpool[(((long)d * 2 + k) * Ireal + i) * O + o];
  }
  W[idx] = (_Float16)acc;
}

// bias_eff[n][o] = emb[n,:] @ b_pool
__global__ void beff_kernel(const float* __restrict__ emb,
                            const float* __restrict__ bpool,
                            float* __restrict__ bias, int O) {
  const int idx = blockIdx.x * 256 + threadIdx.x;
  if (idx >= NN * O) return;
  const int o = idx % O, n = idx / O;
  float acc = 0.f;
#pragma unroll
  for (int d = 0; d < DD; ++d) acc += emb[n * DD + d] * bpool[d * O + o];
  bias[idx] = acc;
}

// ---------------------------------------------------------------------------
// Per-step kernels
// ---------------------------------------------------------------------------

// Build gate input xg[b][n][0:Ipad] = [x_t, h] (f16, zero pad), and seed the
// candidate buffer xg2 with x_t and pad zeros (z*h cols filled by gate_kernel).
__global__ void build_inp_kernel(const float* __restrict__ xsrc,
                                 const float* __restrict__ h,
                                 _Float16* __restrict__ xg,
                                 _Float16* __restrict__ xg2,
                                 int t, int Fin, int Ipad) {
  const int idx = blockIdx.x * 256 + threadIdx.x;
  const int total = BB * NN * Ipad;
  if (idx >= total) return;
  const int i  = idx % Ipad;
  const int bn = idx / Ipad;
  const int n  = bn % NN;
  const int b  = bn / NN;
  const int KI = 2 * Ipad;
  _Float16 v;
  if (i < Fin) {
    v = (_Float16)xsrc[((long)(b * TT + t) * NN + n) * Fin + i];
    xg2[(long)bn * KI + i] = v;
  } else if (i < Fin + HH) {
    v = (_Float16)h[(long)bn * HH + (i - Fin)];
  } else {
    v = (_Float16)0.f;
    xg2[(long)bn * KI + i] = (_Float16)0.f;
  }
  xg[(long)bn * KI + i] = v;
}

// Diffusion GEMM: xg[b][:, Ipad:2*Ipad] = A(2048x2048,f16) @ xg[b][:, 0:Ipad].
// Block: 256 thr = 8 waves; 128-row x 16-col output tile; K staged 64 deep.
// B tile stored TRANSPOSED in LDS so each lane's fragment slice is one
// contiguous 32B read.
__global__ void diff_kernel(const _Float16* __restrict__ A,
                            _Float16* __restrict__ xg, int Ipad) {
  __shared__ __attribute__((aligned(32))) _Float16 sA[128 * 64];
  __shared__ __attribute__((aligned(32))) _Float16 sBT[16 * 64];  // [col][k]
  const int tid  = threadIdx.x;
  const int lane = tid & 31;
  const int wave = tid >> 5;
  const int row0 = blockIdx.x * 128;
  const int col0 = blockIdx.y * 16;
  const int b    = blockIdx.z;
  const int KI   = 2 * Ipad;
  const long xgb = (long)b * NN * KI;
  v8f acc = {};
  for (int kc = 0; kc < NN; kc += 64) {
    // stage A[row0:+128, kc:+64] row-major
#pragma unroll
    for (int g = 0; g < 4; ++g) {
      const int idx = tid + g * 256;            // 0..1023
      const int r = idx >> 3, c8 = (idx & 7) * 8;
      *(uint4*)&sA[r * 64 + c8] =
          *(const uint4*)&A[(long)(row0 + r) * NN + kc + c8];
    }
    if (kc + 64 < NN)
      __builtin_prefetch(&A[(long)(row0 + (tid >> 1)) * NN + kc + 64], 0, 1);
    // stage inp[kc:+64, col0:+16] transposed -> sBT[c][k]
    if (tid < 128) {
      const int k = tid >> 1, c8 = (tid & 1) * 8;
      uint4 tmp = *(const uint4*)&xg[xgb + (long)(kc + k) * KI + col0 + c8];
      const _Float16* hv = (const _Float16*)&tmp;
#pragma unroll
      for (int j = 0; j < 8; ++j) sBT[(c8 + j) * 64 + k] = hv[j];
    }
    __syncthreads();
#pragma unroll
    for (int kk = 0; kk < 2; ++kk) {
      v16h af = frag_a(&sA[wave * 16 * 64 + kk * 32], 64);
      v16h bf = *(const v16h*)&sBT[(lane & 15) * 64 + kk * 32 + (lane >> 4) * 16];
      acc = wmma_f16(af, bf, acc);
    }
    __syncthreads();
  }
  const int np = lane & 15, mb = (lane >> 4) * 8;
#pragma unroll
  for (int r = 0; r < 8; ++r) {
    const int row = row0 + wave * 16 + mb + r;
    xg[xgb + (long)row * KI + Ipad + col0 + np] = (_Float16)acc[r];
  }
}

// Per-node gate: z_r = sigmoid(X_n @ Wg_n + b_n); writes r to rbuf and the
// candidate input cols xg2[:, Fin:Fin+H] = z * h_prev.
// Block = 256 thr (8 waves), one node per block, wave w -> cols [16w,16w+16).
// Weights arrive fragment-packed: one 32B load per lane per k-chunk.
__global__ void gate_kernel(const _Float16* __restrict__ xg,
                            const _Float16* __restrict__ W,
                            const float* __restrict__ bias,
                            const float* __restrict__ hprev,
                            _Float16* __restrict__ xg2,
                            float* __restrict__ rbuf,
                            int Ipad, int Fin) {
  __shared__ __attribute__((aligned(16))) _Float16 sX[8 * 256];
  const int n = blockIdx.x;
  const int tid = threadIdx.x, lane = tid & 31, wave = tid >> 5;
  const int KI = 2 * Ipad;
  const int cpr = KI / 8;                        // 16B chunks per row
  for (int g = tid; g < KI; g += 256) {          // 8 rows * cpr chunks == KI
    const int bb = g / cpr, ch = g % cpr;
    *(uint4*)&sX[bb * KI + ch * 8] =
        *(const uint4*)&xg[((long)bb * NN + n) * KI + ch * 8];
  }
  __syncthreads();
  // packed: fragment (kc, ct=wave) at n*KI*128 + (kc*8 + wave)*512, lane slice *16
  const _Float16* wp = W + (long)n * KI * 128 + (long)wave * 512 + (long)lane * 16;
  v8f acc = {};
  for (int k0 = 0; k0 < KI; k0 += 32) {
    v16h af = frag_a_zpad8(&sX[k0], KI);
    v16h bf = *(const v16h*)wp;
    acc = wmma_f16(af, bf, acc);
    wp += 8 * 512;                               // nct = 8 fragments per k-chunk
  }
  const int np = lane & 15, mb = (lane >> 4) * 8;
  const int o = wave * 16 + np;
  const float bval = bias[n * (2 * HH) + o];
#pragma unroll
  for (int r = 0; r < 8; ++r) {
    const int b = mb + r;
    if (b < BB) {
      const float sig = 1.0f / (1.0f + expf(-(acc[r] + bval)));
      const long bn = (long)b * NN + n;
      if (o < HH) {
        xg2[bn * KI + Fin + o] = (_Float16)(sig * hprev[bn * HH + o]);
      } else {
        rbuf[bn * HH + (o - HH)] = sig;
      }
    }
  }
}

// Per-node update: hc = tanh(Xc_n @ Wu_n + b_n); h = r*h + (1-r)*hc.
// Block = 128 thr (4 waves), one node per block.
__global__ void update_kernel(const _Float16* __restrict__ xg2,
                              const _Float16* __restrict__ W,
                              const float* __restrict__ bias,
                              float* __restrict__ h,
                              const float* __restrict__ rbuf,
                              float* __restrict__ seqdst,
                              int Ipad, int t) {
  __shared__ __attribute__((aligned(16))) _Float16 sX[8 * 256];
  const int n = blockIdx.x;
  const int tid = threadIdx.x, lane = tid & 31, wave = tid >> 5;
  const int KI = 2 * Ipad;
  const int cpr = KI / 8;
  for (int g = tid; g < KI; g += 128) {
    const int bb = g / cpr, ch = g % cpr;
    *(uint4*)&sX[bb * KI + ch * 8] =
        *(const uint4*)&xg2[((long)bb * NN + n) * KI + ch * 8];
  }
  __syncthreads();
  const _Float16* wp = W + (long)n * KI * 64 + (long)wave * 512 + (long)lane * 16;
  v8f acc = {};
  for (int k0 = 0; k0 < KI; k0 += 32) {
    v16h af = frag_a_zpad8(&sX[k0], KI);
    v16h bf = *(const v16h*)wp;
    acc = wmma_f16(af, bf, acc);
    wp += 4 * 512;                               // nct = 4 fragments per k-chunk
  }
  const int np = lane & 15, mb = (lane >> 4) * 8;
  const int o = wave * 16 + np;
  const float bval = bias[n * HH + o];
#pragma unroll
  for (int r = 0; r < 8; ++r) {
    const int b = mb + r;
    if (b < BB) {
      const long bn = (long)b * NN + n;
      const float hc = tanhf(acc[r] + bval);
      const float rr = rbuf[bn * HH + o];
      const float hp = h[bn * HH + o];
      const float hn = rr * hp + (1.0f - rr) * hc;
      h[bn * HH + o] = hn;
      seqdst[(((long)b * TT + t) * NN + n) * HH + o] = hn;
    }
  }
}

__global__ void copy_h_kernel(const float* __restrict__ h, float* __restrict__ dst) {
  const long idx = (long)blockIdx.x * 256 + threadIdx.x;
  if (idx < (long)2 * BB * NN * HH) dst[idx] = h[idx];
}

// ---------------------------------------------------------------------------

extern "C" void kernel_launch(void* const* d_in, const int* in_sizes, int n_in,
                              void* d_out, int out_size, void* d_ws, size_t ws_size,
                              hipStream_t stream) {
  (void)in_sizes; (void)n_in; (void)out_size; (void)ws_size;
  const float* x    = (const float*)d_in[0];
  const float* ist  = (const float*)d_in[1];
  const float* emb  = (const float*)d_in[2];
  const float* g0w  = (const float*)d_in[6];
  const float* g0b  = (const float*)d_in[7];
  const float* u0w  = (const float*)d_in[8];
  const float* u0b  = (const float*)d_in[9];
  const float* g1w  = (const float*)d_in[10];
  const float* g1b  = (const float*)d_in[11];
  const float* u1w  = (const float*)d_in[12];
  const float* u1b  = (const float*)d_in[13];
  float* out = (float*)d_out;

  char* ws = (char*)d_ws;
  size_t off = 0;
  auto take = [&](size_t bytes) -> char* {
    char* p = ws + off;
    off += (bytes + 255) & ~(size_t)255;
    return p;
  };

  // layer geometry: layer0 I=66 -> Ipad 80 (KI 160); layer1 I=128 (KI 256)
  _Float16* A16 = (_Float16*)take((size_t)NN * NN * 2);
  _Float16* Wg0 = (_Float16*)take((size_t)NN * 160 * 128 * 2);
  _Float16* Wu0 = (_Float16*)take((size_t)NN * 160 * 64 * 2);
  _Float16* Wg1 = (_Float16*)take((size_t)NN * 256 * 128 * 2);
  _Float16* Wu1 = (_Float16*)take((size_t)NN * 256 * 64 * 2);
  float* Bg0 = (float*)take((size_t)NN * 128 * 4);
  float* Bu0 = (float*)take((size_t)NN * 64 * 4);
  float* Bg1 = (float*)take((size_t)NN * 128 * 4);
  float* Bu1 = (float*)take((size_t)NN * 64 * 4);
  _Float16* xgA = (_Float16*)take((size_t)BB * NN * 256 * 2);
  _Float16* xgB = (_Float16*)take((size_t)BB * NN * 256 * 2);
  float* rbuf = (float*)take((size_t)BB * NN * HH * 4);
  float* hbuf = (float*)take((size_t)2 * BB * NN * HH * 4);
  float* seq0 = (float*)take((size_t)BB * TT * NN * HH * 4);

  // ---- setup ----
  compute_A_kernel<<<NN, 256, 0, stream>>>(emb, A16);
  {
    long tg = (long)NN * 160 * 128;
    weff_kernel<<<(tg + 255) / 256, 256, 0, stream>>>(emb, g0w, Wg0, 66, 80, 128, 160);
    long tu = (long)NN * 160 * 64;
    weff_kernel<<<(tu + 255) / 256, 256, 0, stream>>>(emb, u0w, Wu0, 66, 80, 64, 160);
    long tg1 = (long)NN * 256 * 128;
    weff_kernel<<<(tg1 + 255) / 256, 256, 0, stream>>>(emb, g1w, Wg1, 128, 128, 128, 256);
    long tu1 = (long)NN * 256 * 64;
    weff_kernel<<<(tu1 + 255) / 256, 256, 0, stream>>>(emb, u1w, Wu1, 128, 128, 64, 256);
  }
  beff_kernel<<<(NN * 128 + 255) / 256, 256, 0, stream>>>(emb, g0b, Bg0, 128);
  beff_kernel<<<(NN * 64 + 255) / 256, 256, 0, stream>>>(emb, u0b, Bu0, 64);
  beff_kernel<<<(NN * 128 + 255) / 256, 256, 0, stream>>>(emb, g1b, Bg1, 128);
  beff_kernel<<<(NN * 64 + 255) / 256, 256, 0, stream>>>(emb, u1b, Bu1, 64);
  hipMemcpyAsync(hbuf, ist, (size_t)2 * BB * NN * HH * 4,
                 hipMemcpyDeviceToDevice, stream);

  float* h0 = hbuf;
  float* h1 = hbuf + (size_t)BB * NN * HH;

  // ---- recurrence ----
  for (int t = 0; t < TT; ++t) {
    // layer 0: Fin=2, Ipad=80
    {
      const int total = BB * NN * 80;
      build_inp_kernel<<<(total + 255) / 256, 256, 0, stream>>>(x, h0, xgA, xgB, t, 2, 80);
      diff_kernel<<<dim3(NN / 128, 80 / 16, BB), 256, 0, stream>>>(A16, xgA, 80);
      gate_kernel<<<NN, 256, 0, stream>>>(xgA, Wg0, Bg0, h0, xgB, rbuf, 80, 2);
      diff_kernel<<<dim3(NN / 128, 80 / 16, BB), 256, 0, stream>>>(A16, xgB, 80);
      update_kernel<<<NN, 128, 0, stream>>>(xgB, Wu0, Bu0, h0, rbuf, seq0, 80, t);
    }
    // layer 1: Fin=64, Ipad=128
    {
      const int total = BB * NN * 128;
      build_inp_kernel<<<(total + 255) / 256, 256, 0, stream>>>(seq0, h1, xgA, xgB, t, 64, 128);
      diff_kernel<<<dim3(NN / 128, 128 / 16, BB), 256, 0, stream>>>(A16, xgA, 128);
      gate_kernel<<<NN, 256, 0, stream>>>(xgA, Wg1, Bg1, h1, xgB, rbuf, 128, 64);
      diff_kernel<<<dim3(NN / 128, 128 / 16, BB), 256, 0, stream>>>(A16, xgB, 128);
      update_kernel<<<NN, 128, 0, stream>>>(xgB, Wu1, Bu1, h1, rbuf, out, 128, t);
    }
  }

  const long hn = (long)2 * BB * NN * HH;
  copy_h_kernel<<<(hn + 255) / 256, 256, 0, stream>>>(hbuf, out + (long)BB * TT * NN * HH);
}
